// NodeEncoder_46772193853584
// MI455X (gfx1250) — matrix operand order repaired and checked by reading, not compiled, not run on hardware
//
#include <hip/hip_runtime.h>
#include <hip/hip_bf16.h>

typedef __attribute__((ext_vector_type(16))) _Float16 v16h;
typedef __attribute__((ext_vector_type(8)))  float    v8f;

#define NEG_SLOPE 0.2f
#define ALPHA_F 0.1f
#define BN_EPS 1e-5f

__device__ inline v8f wmma_f16(v16h a, v16h b, v8f c) {
    return __builtin_amdgcn_wmma_f32_16x16x32_f16(false, a, false, b, (short)0, c, false, false);
}
// ISA 7.12.2: 16-bit A 16x32 layout. half index j, lane -> K
__device__ inline int kmapA(int j, int lane) {
    return (j & 7) + ((j & 8) ? 16 : 0) + ((lane & 16) ? 8 : 0);
}
// B 32x16: same K order as A (lanes 0-15 K=0..7,16..23; lanes 16-31 K=8..15,24..31)
__device__ inline int kmapB(int j, int lane) { return (j & 7) + ((j & 8) ? 16 : 0) + ((lane & 16) ? 8 : 0); }

// ---------------- small utility kernels ----------------
// convert f32 [L][K][Nc] -> f16 transposed [L][Nc][K]
__global__ void f2h_tr_kernel(const float* __restrict__ in, _Float16* __restrict__ out,
                              int K, int Nc, int total) {
    int i = blockIdx.x * blockDim.x + threadIdx.x;
    if (i >= total) return;
    int per = K * Nc;
    int l = i / per, rem = i - l * per;
    int n = rem / K, k = rem - n * K;
    out[i] = (_Float16)in[l * per + k * Nc + n];
}

__global__ void lin0_kernel(const float* __restrict__ X, const float* __restrict__ J,
                            const float* __restrict__ W, const float* __restrict__ B,
                            float* __restrict__ H, int M) {
    int i = blockIdx.x * blockDim.x + threadIdx.x;
    if (i >= M * 128) return;
    int n = i >> 7, d = i & 127;
    H[i] = X[n * 2 + 0] * W[d] + X[n * 2 + 1] * W[128 + d] + J[n] * W[256 + d] + B[d];
}

__global__ void count_kernel(const int* __restrict__ ei, int* __restrict__ counts, int E_, int N_) {
    int i = blockIdx.x * blockDim.x + threadIdx.x;
    if (i >= E_ + N_) return;
    int dst = (i < E_) ? ei[E_ + i] : (i - E_);
    atomicAdd(&counts[dst], 1);
}

__global__ void scan_kernel(const int* __restrict__ cnt, int* __restrict__ offs, int n) {
    __shared__ int buf[256];
    __shared__ int carry;
    int t = threadIdx.x;
    if (t == 0) carry = 0;
    __syncthreads();
    for (int base = 0; base < n; base += 256) {
        int i = base + t;
        int v = (i < n) ? cnt[i] : 0;
        buf[t] = v;
        __syncthreads();
        for (int o = 1; o < 256; o <<= 1) {
            int add = (t >= o) ? buf[t - o] : 0;
            __syncthreads();
            buf[t] += add;
            __syncthreads();
        }
        if (i < n) offs[i] = carry + buf[t] - v;
        __syncthreads();
        if (t == 0) carry += buf[255];
        __syncthreads();
    }
    if (t == 0) offs[n] = carry;
}

__global__ void fill_kernel(const int* __restrict__ ei, int* __restrict__ cursor,
                            int* __restrict__ csrc, int E_, int N_) {
    int i = blockIdx.x * blockDim.x + threadIdx.x;
    if (i >= E_ + N_) return;
    int src, dst;
    if (i < E_) { src = ei[i]; dst = ei[E_ + i]; }
    else        { src = i - E_; dst = i - E_; }
    int pos = atomicAdd(&cursor[dst], 1);
    csrc[pos] = src;
}

__global__ void dinv_kernel(const int* __restrict__ offs, float* __restrict__ dinv, int N_) {
    int n = blockIdx.x * blockDim.x + threadIdx.x;
    if (n >= N_) return;
    int deg = offs[n + 1] - offs[n];
    if (deg < 1) deg = 1;
    dinv[n] = rsqrtf((float)deg);
}

// ---------------- WMMA GEMM: xl = h @ W (K=128, N=128, no bias) ----------------
// W is f16, transposed: W[n][k]
__global__ __launch_bounds__(256) void gemm_gat(const float* __restrict__ A,
                                                const _Float16* __restrict__ W,
                                                float* __restrict__ Out, int M) {
    __shared__ __align__(16) _Float16 As[32][130];
    __shared__ __align__(16) _Float16 Bs[128][40];   // [col][k], 80B row stride
    int tid = threadIdx.x, lane = tid & 31, wv = tid >> 5;
    int rowbase = blockIdx.x * 32;
    for (int i = tid; i < 32 * 128; i += 256) {
        int r = i >> 7, k = i & 127;
        int g = rowbase + r;
        As[r][k] = (g < M) ? (_Float16)A[(size_t)g * 128 + k] : (_Float16)0.f;
    }
    int rs = wv & 1, cp = wv >> 1, m = lane & 15;
    v8f acc0 = {}, acc1 = {};
    for (int k0 = 0; k0 < 128; k0 += 32) {
        __syncthreads();
        {
            int lc = tid >> 1, kk = (tid & 1) * 16;
            const uint4* g = (const uint4*)(W + (size_t)lc * 128 + k0 + kk);
            uint4* s = (uint4*)&Bs[lc][kk];
            s[0] = g[0]; s[1] = g[1];
        }
        __syncthreads();
        v16h a;
#pragma unroll
        for (int j = 0; j < 16; ++j) a[j] = As[rs * 16 + m][k0 + kmapA(j, lane)];
        v16h b0, b1;
#pragma unroll
        for (int j = 0; j < 16; ++j) {
            int kl = kmapB(j, lane);
            b0[j] = Bs[(cp * 2 + 0) * 16 + m][kl];
            b1[j] = Bs[(cp * 2 + 1) * 16 + m][kl];
        }
        acc0 = wmma_f16(a, b0, acc0);
        acc1 = wmma_f16(a, b1, acc1);
    }
    int mhi = (lane & 16) ? 8 : 0;
#pragma unroll
    for (int r = 0; r < 8; ++r) {
        int grow = rowbase + rs * 16 + r + mhi;
        if (grow < M) {
            Out[(size_t)grow * 128 + (cp * 2 + 0) * 16 + m] = acc0[r];
            Out[(size_t)grow * 128 + (cp * 2 + 1) * 16 + m] = acc1[r];
        }
    }
}

// ---------------- attention coefficients ----------------
__global__ void attn_kernel(const float* __restrict__ XL, const float* __restrict__ asrc,
                            const float* __restrict__ adst, float* __restrict__ AS,
                            float* __restrict__ AD, int M) {
    int i = blockIdx.x * blockDim.x + threadIdx.x;
    if (i >= M * 8) return;
    int n = i >> 3, h = i & 7;
    float s1 = 0.f, s2 = 0.f;
#pragma unroll
    for (int v = 0; v < 16; ++v) {
        float x = XL[(size_t)n * 128 + h * 16 + v];
        s1 += x * asrc[h * 16 + v];
        s2 += x * adst[h * 16 + v];
    }
    AS[i] = s1; AD[i] = s2;
}

// ---------------- GAT softmax-aggregate (CSR gather, streaming softmax) ----------------
__global__ __launch_bounds__(128) void gat_aggr_kernel(const float* __restrict__ XL,
                                                       const float* __restrict__ AS,
                                                       const float* __restrict__ AD,
                                                       const int* __restrict__ offs,
                                                       const int* __restrict__ csrc,
                                                       float* __restrict__ OUT) {
    int n = blockIdx.x, t = threadIdx.x;
    int beg = offs[n], end = offs[n + 1];
    __shared__ float mS[8], sS[8], adS[8];
    if (t < 8) {
        float ad = AD[n * 8 + t];
        float m = -1e30f, s = 0.f;
        for (int j = beg; j < end; ++j) {
            int sj = csrc[j];
            float e = AS[sj * 8 + t] + ad;
            e = (e > 0.f) ? e : NEG_SLOPE * e;
            float nm = fmaxf(m, e);
            s = s * __expf(m - nm) + __expf(e - nm);
            m = nm;
        }
        mS[t] = m; sS[t] = s; adS[t] = ad;
    }
    __syncthreads();
    int h = t >> 4;
    float m = mS[h], inv = 1.f / sS[h], ad = adS[h];
    float acc = 0.f;
    for (int j = beg; j < end; ++j) {
        int sj = csrc[j];
        float e = AS[sj * 8 + h] + ad;
        e = (e > 0.f) ? e : NEG_SLOPE * e;
        acc += __expf(e - m) * inv * XL[(size_t)sj * 128 + t];
    }
    OUT[(size_t)n * 128 + t] = acc;
}

// ---------------- z1 = h + (sum_h gat) @ lin1, accumulate BN col stats ----------------
__global__ __launch_bounds__(128) void z1_kernel(const float* __restrict__ H,
                                                 const float* __restrict__ G,
                                                 const float* __restrict__ lin1,
                                                 float* __restrict__ Z,
                                                 float* __restrict__ colsum,
                                                 float* __restrict__ colsq, int M) {
    __shared__ float gh[16];
    int d = threadIdx.x;
    float w[16];
#pragma unroll
    for (int c = 0; c < 16; ++c) w[c] = lin1[c * 128 + d];
    float ls = 0.f, lq = 0.f;
    for (int row = blockIdx.x; row < M; row += gridDim.x) {
        if (d < 16) {
            float s = 0.f;
#pragma unroll
            for (int h8 = 0; h8 < 8; ++h8) s += G[(size_t)row * 128 + h8 * 16 + d];
            gh[d] = s;
        }
        __syncthreads();
        float z = H[(size_t)row * 128 + d];
#pragma unroll
        for (int c = 0; c < 16; ++c) z += gh[c] * w[c];
        Z[(size_t)row * 128 + d] = z;
        ls += z; lq += z * z;
        __syncthreads();
    }
    atomicAdd(&colsum[d], ls);
    atomicAdd(&colsq[d], lq);
}

__global__ void bn_finalize(const float* __restrict__ colsum, const float* __restrict__ colsq,
                            const float* __restrict__ g, const float* __restrict__ b,
                            float* __restrict__ scale, float* __restrict__ shift, int M) {
    int d = threadIdx.x;
    float mu = colsum[d] / (float)M;
    float var = colsq[d] / (float)M - mu * mu;
    float sc = g[d] * rsqrtf(var + BN_EPS);
    scale[d] = sc;
    shift[d] = b[d] - mu * sc;
}

__global__ void bn_apply(const float* __restrict__ in, const float* __restrict__ scale,
                         const float* __restrict__ shift, float* __restrict__ out, int total) {
    int i = blockIdx.x * blockDim.x + threadIdx.x;
    if (i >= total) return;
    int d = i & 127;
    out[i] = in[i] * scale[d] + shift[d];
}

// ---------------- fused MLP: z2 = relu(h1@W2+b2)@W3+b3 + h1, plus BN col stats ----------------
// W2 f16 transposed [512][128]; W3 f16 transposed [128][512]
__global__ __launch_bounds__(256) void mlp_fused(const float* __restrict__ H1,
                                                 const _Float16* __restrict__ W2,
                                                 const float* __restrict__ B2,
                                                 const _Float16* __restrict__ W3,
                                                 const float* __restrict__ B3,
                                                 float* __restrict__ Z2,
                                                 float* __restrict__ colsum,
                                                 float* __restrict__ colsq, int M) {
    __shared__ __align__(16) _Float16 As[32][130];
    __shared__ _Float16 T[32][516];
    __shared__ __align__(16) _Float16 Bs[128][40];   // [col][k]
    __shared__ float csum[128], csq[128];
    int tid = threadIdx.x, lane = tid & 31, wv = tid >> 5;
    if (tid < 128) { csum[tid] = 0.f; csq[tid] = 0.f; }
    int rowbase = blockIdx.x * 32;
    for (int i = tid; i < 32 * 128; i += 256) {
        int r = i >> 7, k = i & 127;
        int g = rowbase + r;
        As[r][k] = (g < M) ? (_Float16)H1[(size_t)g * 128 + k] : (_Float16)0.f;
    }
    int rs = wv & 1, cp = wv >> 1, m = lane & 15, mhi = (lane & 16) ? 8 : 0;

    // stage 1: T = relu(h1 @ W2 + b2), 4 column groups of 128
    for (int cg = 0; cg < 4; ++cg) {
        v8f a0 = {}, a1 = {};
        for (int k0 = 0; k0 < 128; k0 += 32) {
            __syncthreads();
            {
                int lc = tid >> 1, kk = (tid & 1) * 16;
                const uint4* g = (const uint4*)(W2 + ((size_t)(cg * 128 + lc)) * 128 + k0 + kk);
                uint4* s = (uint4*)&Bs[lc][kk];
                s[0] = g[0]; s[1] = g[1];
            }
            __syncthreads();
            v16h a;
#pragma unroll
            for (int j = 0; j < 16; ++j) a[j] = As[rs * 16 + m][k0 + kmapA(j, lane)];
            v16h b0, b1;
#pragma unroll
            for (int j = 0; j < 16; ++j) {
                int kl = kmapB(j, lane);
                b0[j] = Bs[(cp * 2 + 0) * 16 + m][kl];
                b1[j] = Bs[(cp * 2 + 1) * 16 + m][kl];
            }
            a0 = wmma_f16(a, b0, a0);
            a1 = wmma_f16(a, b1, a1);
        }
#pragma unroll
        for (int r = 0; r < 8; ++r) {
            int mr = rs * 16 + r + mhi;
            int c0 = cg * 128 + (cp * 2 + 0) * 16 + m;
            int c1 = cg * 128 + (cp * 2 + 1) * 16 + m;
            float v0 = a0[r] + B2[c0];
            float v1 = a1[r] + B2[c1];
            T[mr][c0] = (_Float16)fmaxf(v0, 0.f);
            T[mr][c1] = (_Float16)fmaxf(v1, 0.f);
        }
    }
    __syncthreads();

    // stage 2: z2 = T @ W3 + b3 + h1
    v8f z0 = {}, z1v = {};
    for (int k0 = 0; k0 < 512; k0 += 32) {
        __syncthreads();
        {
            int lc = tid >> 1, kk = (tid & 1) * 16;
            const uint4* g = (const uint4*)(W3 + (size_t)lc * 512 + k0 + kk);
            uint4* s = (uint4*)&Bs[lc][kk];
            s[0] = g[0]; s[1] = g[1];
        }
        __syncthreads();
        v16h a;
#pragma unroll
        for (int j = 0; j < 16; ++j) a[j] = T[rs * 16 + m][k0 + kmapA(j, lane)];
        v16h b0, b1;
#pragma unroll
        for (int j = 0; j < 16; ++j) {
            int kl = kmapB(j, lane);
            b0[j] = Bs[(cp * 2 + 0) * 16 + m][kl];
            b1[j] = Bs[(cp * 2 + 1) * 16 + m][kl];
        }
        z0 = wmma_f16(a, b0, z0);
        z1v = wmma_f16(a, b1, z1v);
    }
#pragma unroll
    for (int r = 0; r < 8; ++r) {
        int mr = rs * 16 + r + mhi;
        int grow = rowbase + mr;
        if (grow < M) {
            int c0 = (cp * 2 + 0) * 16 + m;
            int c1 = (cp * 2 + 1) * 16 + m;
            float v0 = z0[r] + B3[c0] + H1[(size_t)grow * 128 + c0];
            float v1 = z1v[r] + B3[c1] + H1[(size_t)grow * 128 + c1];
            Z2[(size_t)grow * 128 + c0] = v0;
            Z2[(size_t)grow * 128 + c1] = v1;
            atomicAdd(&csum[c0], v0); atomicAdd(&csq[c0], v0 * v0);
            atomicAdd(&csum[c1], v1); atomicAdd(&csq[c1], v1 * v1);
        }
    }
    __syncthreads();
    if (tid < 128) {
        atomicAdd(&colsum[tid], csum[tid]);
        atomicAdd(&colsq[tid], csq[tid]);
    }
}

// ---------------- APPNP step (CSR gather) ----------------
__global__ __launch_bounds__(128) void appnp_kernel(const float* __restrict__ cur,
                                                    const float* __restrict__ h0,
                                                    const int* __restrict__ offs,
                                                    const int* __restrict__ csrc,
                                                    const float* __restrict__ dinv,
                                                    float* __restrict__ nxt) {
    int n = blockIdx.x, t = threadIdx.x;
    int beg = offs[n], end = offs[n + 1];
    float acc = 0.f;
    for (int j = beg; j < end; ++j) {
        int s = csrc[j];
        acc += dinv[s] * cur[(size_t)s * 128 + t];
    }
    nxt[(size_t)n * 128 + t] = (1.f - ALPHA_F) * dinv[n] * acc + ALPHA_F * h0[(size_t)n * 128 + t];
}

__global__ void out_kernel(const float* __restrict__ H, const float* __restrict__ J,
                           float* __restrict__ out, int N_) {
    int i = blockIdx.x * blockDim.x + threadIdx.x;
    if (i >= N_ * 129) return;
    int n = i / 129, d = i - n * 129;
    out[i] = (d < 128) ? H[(size_t)n * 128 + d] : J[n];
}

static inline int ceil_div(int a, int b) { return (a + b - 1) / b; }

extern "C" void kernel_launch(void* const* d_in, const int* in_sizes, int n_in,
                              void* d_out, int out_size, void* d_ws, size_t ws_size,
                              hipStream_t stream) {
    const float* x      = (const float*)d_in[0];
    const float* J      = (const float*)d_in[1];
    const int*   ei     = (const int*)d_in[2];
    const float* lin0_w = (const float*)d_in[3];
    const float* lin0_b = (const float*)d_in[4];
    const float* gat_w  = (const float*)d_in[5];
    const float* att_s  = (const float*)d_in[6];
    const float* att_d  = (const float*)d_in[7];
    const float* lin1_w = (const float*)d_in[8];
    const float* bn1_g  = (const float*)d_in[9];
    const float* bn1_b  = (const float*)d_in[10];
    const float* lin2_w = (const float*)d_in[11];
    const float* lin2_b = (const float*)d_in[12];
    const float* lin3_w = (const float*)d_in[13];
    const float* lin3_b = (const float*)d_in[14];
    const float* bn2_g  = (const float*)d_in[15];
    const float* bn2_b  = (const float*)d_in[16];

    const int Nn = in_sizes[1];          // 50000 nodes (J is [N])
    const int Ee = in_sizes[2] / 2;      // 800000 edges
    const int EP = Ee + Nn;              // with self-loops
    const int L = 2;

    // workspace carving, 256B-aligned
    char* base = (char*)d_ws;
    size_t off = 0;
    auto carve = [&](size_t bytes) -> void* {
        void* r = base + off;
        off = (off + bytes + 255) & ~(size_t)255;
        return r;
    };
    float* bh   = (float*)carve((size_t)Nn * 128 * 4);
    float* bh1  = (float*)carve((size_t)Nn * 128 * 4);
    float* bxl  = (float*)carve((size_t)Nn * 128 * 4);   // xl / z1 / z2
    float* bgat = (float*)carve((size_t)Nn * 128 * 4);   // gat out, later APPNP h0
    float* bas  = (float*)carve((size_t)Nn * 8 * 4);
    float* bad  = (float*)carve((size_t)Nn * 8 * 4);
    int*   counts = (int*)carve((size_t)Nn * 4);
    int*   offs   = (int*)carve((size_t)(Nn + 1) * 4);
    int*   cursor = (int*)carve((size_t)Nn * 4);
    int*   csrc   = (int*)carve((size_t)EP * 4);
    float* dinv   = (float*)carve((size_t)Nn * 4);
    float* stats  = (float*)carve(256 * 4);  // csum[128] | csq[128]
    float* csum = stats, *csq = stats + 128;
    float* ss     = (float*)carve(256 * 4);  // scale[128] | shift[128]
    float* scale = ss, *shift = ss + 128;
    _Float16* gw16 = (_Float16*)carve((size_t)L * 128 * 128 * 2);  // transposed [n][k]
    _Float16* w216 = (_Float16*)carve((size_t)L * 512 * 128 * 2);  // transposed [512][128]
    _Float16* w316 = (_Float16*)carve((size_t)L * 128 * 512 * 2);  // transposed [128][512]

    // ---- convert + transpose weights to f16 [Nc][K] ----
    f2h_tr_kernel<<<ceil_div(L * 128 * 128, 256), 256, 0, stream>>>(gat_w, gw16, 128, 128, L * 128 * 128);
    f2h_tr_kernel<<<ceil_div(L * 128 * 512, 256), 256, 0, stream>>>(lin2_w, w216, 128, 512, L * 128 * 512);
    f2h_tr_kernel<<<ceil_div(L * 512 * 128, 256), 256, 0, stream>>>(lin3_w, w316, 512, 128, L * 512 * 128);

    // ---- build CSR by destination (incl. self-loops) ----
    hipMemsetAsync(counts, 0, (size_t)Nn * 4, stream);
    count_kernel<<<ceil_div(EP, 256), 256, 0, stream>>>(ei, counts, Ee, Nn);
    scan_kernel<<<1, 256, 0, stream>>>(counts, offs, Nn);
    hipMemcpyAsync(cursor, offs, (size_t)Nn * 4, hipMemcpyDeviceToDevice, stream);
    fill_kernel<<<ceil_div(EP, 256), 256, 0, stream>>>(ei, cursor, csrc, Ee, Nn);
    dinv_kernel<<<ceil_div(Nn, 256), 256, 0, stream>>>(offs, dinv, Nn);

    // ---- input embedding ----
    lin0_kernel<<<ceil_div(Nn * 128, 256), 256, 0, stream>>>(x, J, lin0_w, lin0_b, bh, Nn);

    // ---- attention layers ----
    for (int l = 0; l < L; ++l) {
        gemm_gat<<<ceil_div(Nn, 32), 256, 0, stream>>>(bh, gw16 + (size_t)l * 128 * 128, bxl, Nn);
        attn_kernel<<<ceil_div(Nn * 8, 256), 256, 0, stream>>>(bxl, att_s + l * 128, att_d + l * 128,
                                                               bas, bad, Nn);
        gat_aggr_kernel<<<Nn, 128, 0, stream>>>(bxl, bas, bad, offs, csrc, bgat);

        hipMemsetAsync(stats, 0, 256 * 4, stream);
        z1_kernel<<<1024, 128, 0, stream>>>(bh, bgat, lin1_w + l * 16 * 128, bxl, csum, csq, Nn);
        bn_finalize<<<1, 128, 0, stream>>>(csum, csq, bn1_g + l * 128, bn1_b + l * 128, scale, shift, Nn);
        bn_apply<<<ceil_div(Nn * 128, 256), 256, 0, stream>>>(bxl, scale, shift, bh1, Nn * 128);

        hipMemsetAsync(stats, 0, 256 * 4, stream);
        mlp_fused<<<ceil_div(Nn, 32), 256, 0, stream>>>(bh1, w216 + (size_t)l * 512 * 128,
                                                        lin2_b + l * 512,
                                                        w316 + (size_t)l * 128 * 512,
                                                        lin3_b + l * 128, bxl, csum, csq, Nn);
        bn_finalize<<<1, 128, 0, stream>>>(csum, csq, bn2_g + l * 128, bn2_b + l * 128, scale, shift, Nn);
        bn_apply<<<ceil_div(Nn * 128, 256), 256, 0, stream>>>(bxl, scale, shift, bh, Nn * 128);
    }

    // ---- APPNP: 10 steps, gather form ----
    hipMemcpyAsync(bgat, bh, (size_t)Nn * 128 * 4, hipMemcpyDeviceToDevice, stream);  // h0
    float* cur = bh;
    float* nxt = bh1;
    for (int k = 0; k < 10; ++k) {
        appnp_kernel<<<Nn, 128, 0, stream>>>(cur, bgat, offs, csrc, dinv, nxt);
        float* tmp = cur; cur = nxt; nxt = tmp;
    }

    out_kernel<<<ceil_div(Nn * 129, 256), 256, 0, stream>>>(cur, J, (float*)d_out, Nn);
}
